// ShareAttCrossAttention_29085518528829
// MI455X (gfx1250) — compile-verified, matrix operand order checked by reading, not run on hardware
//
#include <hip/hip_runtime.h>
#include <hip/hip_bf16.h>

// ---------------------------------------------------------------------------
// Shapes (hardcoded from reference): B=4, N=2048, C=512, H=8, Dh=64
// ---------------------------------------------------------------------------
#define BATCH 4
#define SEQ   2048
#define DIMC  512
#define NHEAD 8
#define DHEAD 64

typedef _Float16 half16 __attribute__((ext_vector_type(16)));
typedef _Float16 half8  __attribute__((ext_vector_type(8)));
typedef float    float8 __attribute__((ext_vector_type(8)));

static __device__ __forceinline__ float8 wmma_f16(half16 a, half16 b, float8 c) {
    // D = A(16x32 f16) x B(32x16 f16) + C(16x16 f32)
    return __builtin_amdgcn_wmma_f32_16x16x32_f16(false, a, false, b,
                                                  (short)0, c, false, false);
}

// Async global->LDS copy (CDNA5): per lane, 16B MEM[vaddr] -> LDS[ldsoff].
// Tracked by ASYNCcnt; completion enforced with s_wait_asynccnt.
static __device__ __forceinline__ void async_load_b128(unsigned ldsoff, const void* gaddr) {
    asm volatile("global_load_async_to_lds_b128 %0, %1, off"
                 :: "v"(ldsoff), "v"(gaddr) : "memory");
}
#define WAIT_ASYNCCNT_0() asm volatile("s_wait_asynccnt 0" ::: "memory")
#define WAIT_ASYNCCNT_1() asm volatile("s_wait_asynccnt 1" ::: "memory")
#define WAIT_ASYNCCNT_4() asm volatile("s_wait_asynccnt 4" ::: "memory")

// LDS byte offset of a __shared__ element (flat LDS addr carries offset in low bits)
#define LDS_OFF(p) ((unsigned)(uintptr_t)(void*)(p))

// ---------------------------------------------------------------------------
// Kernel 1: LayerNorm over last dim (512) + cast to f16.  One block per row.
// ---------------------------------------------------------------------------
__global__ __launch_bounds__(256)
void ln_cast_kernel(const float* __restrict__ in, const float* __restrict__ g,
                    const float* __restrict__ be, _Float16* __restrict__ out) {
    const int row = blockIdx.x;
    const int t   = threadIdx.x;                // 256 threads, 2 elems each
    const float* r = in + (size_t)row * DIMC;
    float v0 = r[t], v1 = r[t + 256];

    __shared__ float red[8];
    float s = v0 + v1;
    for (int o = 16; o; o >>= 1) s += __shfl_xor(s, o, 32);
    if ((t & 31) == 0) red[t >> 5] = s;
    __syncthreads();
    s = red[0]+red[1]+red[2]+red[3]+red[4]+red[5]+red[6]+red[7];
    const float mu = s * (1.0f / DIMC);

    const float d0 = v0 - mu, d1 = v1 - mu;
    float q = d0*d0 + d1*d1;
    __syncthreads();
    for (int o = 16; o; o >>= 1) q += __shfl_xor(q, o, 32);
    if ((t & 31) == 0) red[t >> 5] = q;
    __syncthreads();
    q = red[0]+red[1]+red[2]+red[3]+red[4]+red[5]+red[6]+red[7];
    const float rs = rsqrtf(q * (1.0f / DIMC) + 1e-5f);

    out[(size_t)row * DIMC + t]       = (_Float16)(d0 * rs * g[t]       + be[t]);
    out[(size_t)row * DIMC + t + 256] = (_Float16)(d1 * rs * g[t + 256] + be[t + 256]);
}

// ---------------------------------------------------------------------------
// Kernel 2: f32 -> f16 weight convert
// ---------------------------------------------------------------------------
__global__ __launch_bounds__(256)
void cvt_f16_kernel(const float* __restrict__ src, _Float16* __restrict__ dst, int n) {
    int i = blockIdx.x * 256 + threadIdx.x;
    if (i < n) dst[i] = (_Float16)src[i];
}

// ---------------------------------------------------------------------------
// Kernel 3: WMMA GEMM  C[m][n] = sum_k A[m][k] * W[n][k] + bias[n]
//   M = 8192, N = 512, K = 512.  Block tile 64x64x64, 8 waves (4 Mx2 N).
//   Non-scramble path: double-buffered LDS fed by GLOBAL_LOAD_ASYNC_TO_LDS_B128
//   (4 async ops/wave/tile; counted s_wait_asynccnt keeps next tile in flight
//   while current tile is consumed by WMMAs).
//   SCRAMBLE path (o_d.transpose(0,2,1).reshape gather): register pipeline.
// ---------------------------------------------------------------------------
template<bool SCRAMBLE, bool OUTF32>
__global__ __launch_bounds__(256)
void gemm_wmma_kernel(const _Float16* __restrict__ A, const _Float16* __restrict__ W,
                      const float* __restrict__ bias,
                      float* __restrict__ outF, _Float16* __restrict__ outH) {
    __shared__ _Float16 As[2][64][72];   // 64 K + pad (144B stride, 16B aligned)
    __shared__ _Float16 Bs[2][64][72];

    const int m0 = blockIdx.x * 64, n0 = blockIdx.y * 64;
    const int tid = threadIdx.x, wave = tid >> 5, lane = tid & 31;
    const int wm = wave & 3, wn = wave >> 2;       // 4x2 wave grid
    const int ln = lane & 15, kh = (lane >> 4) * 8;

    float8 acc0 = {}, acc1 = {};

    // ---- async staging map: instr covers 4 rows x 64 f16 (32 lanes x 16B) ----
    const int sr = (lane >> 3), sc8 = (lane & 7) * 8;
    auto issueTile = [&](int kk, int buf) {
        #pragma unroll
        for (int p = 0; p < 2; ++p) {
            const int r = wave * 8 + p * 4 + sr;
            async_load_b128(LDS_OFF(&As[buf][r][sc8]),
                            A + (size_t)(m0 + r) * DIMC + kk + sc8);
            async_load_b128(LDS_OFF(&Bs[buf][r][sc8]),
                            W + (size_t)(n0 + r) * DIMC + kk + sc8);
        }
    };

    // ---- scramble register path ----
    const int ar = tid >> 2, ac = (tid & 3) * 8;   // row 0..63, chunks ac / ac+32
    half8 aR0, aR1, bR0, bR1;
    auto loadA = [&](int kk) {
        const int grow = m0 + ar;
        const int b = grow >> 11, p = grow & 2047;
        const size_t base = ((size_t)(b << 11) + ((size_t)(p & 3) << 9) + kk + ac) * DIMC
                          + (size_t)(p >> 2);
        #pragma unroll
        for (int i = 0; i < 8; ++i) {
            aR0[i] = A[base + (size_t)i * DIMC];
            aR1[i] = A[base + (size_t)(i + 32) * DIMC];
        }
    };
    auto loadB = [&](int kk) {
        const _Float16* p = W + (size_t)(n0 + ar) * DIMC + kk + ac;
        bR0 = *(const half8*)p;
        bR1 = *(const half8*)(p + 32);
    };

    if constexpr (!SCRAMBLE) {
        issueTile(0, 0);
    } else {
        loadA(0); loadB(0);
    }

    int cur = 0;
    for (int kk = 0; kk < DIMC; kk += 64) {
        const bool hasNext = (kk + 64 < DIMC);
        if constexpr (!SCRAMBLE) {
            // next tile -> other buffer; wait only for the current tile's 4 ops
            if (hasNext) { issueTile(kk + 64, cur ^ 1); WAIT_ASYNCCNT_4(); }
            else         { WAIT_ASYNCCNT_0(); }
            __syncthreads();
        } else {
            *(half8*)&As[cur][ar][ac]      = aR0;
            *(half8*)&As[cur][ar][ac + 32] = aR1;
            *(half8*)&Bs[cur][ar][ac]      = bR0;
            *(half8*)&Bs[cur][ar][ac + 32] = bR1;
            __syncthreads();
            if (hasNext) { loadA(kk + 64); loadB(kk + 64); }
        }

        // ---- fragments (ISA 16-bit A/B layout: lane=row, klo=(lane>>4)*8) ----
        union { half16 v; half8 h[2]; } u;
        half16 a0, a1, b0lo, b0hi, b1lo, b1hi;
        u.h[0] = *(const half8*)&As[cur][wm*16 + ln][kh];
        u.h[1] = *(const half8*)&As[cur][wm*16 + ln][16 + kh];  a0 = u.v;
        u.h[0] = *(const half8*)&As[cur][wm*16 + ln][32 + kh];
        u.h[1] = *(const half8*)&As[cur][wm*16 + ln][48 + kh];  a1 = u.v;
        u.h[0] = *(const half8*)&Bs[cur][wn*32 + ln][kh];
        u.h[1] = *(const half8*)&Bs[cur][wn*32 + ln][16 + kh];  b0lo = u.v;
        u.h[0] = *(const half8*)&Bs[cur][wn*32 + ln][32 + kh];
        u.h[1] = *(const half8*)&Bs[cur][wn*32 + ln][48 + kh];  b0hi = u.v;
        u.h[0] = *(const half8*)&Bs[cur][wn*32 + 16 + ln][kh];
        u.h[1] = *(const half8*)&Bs[cur][wn*32 + 16 + ln][16 + kh];  b1lo = u.v;
        u.h[0] = *(const half8*)&Bs[cur][wn*32 + 16 + ln][32 + kh];
        u.h[1] = *(const half8*)&Bs[cur][wn*32 + 16 + ln][48 + kh];  b1hi = u.v;

        acc0 = wmma_f16(a0, b0lo, acc0);
        acc0 = wmma_f16(a1, b0hi, acc0);
        acc1 = wmma_f16(a0, b1lo, acc1);
        acc1 = wmma_f16(a1, b1hi, acc1);
        __syncthreads();   // all reads of buf[cur] done before it is re-targeted
        cur ^= 1;
    }

    // ---- epilogue: C layout = VGPR r -> row r (lanes 0-15) / r+8 (16-31) ----
    const int rbase = m0 + wm*16 + ((lane < 16) ? 0 : 8);
    const int c0 = n0 + wn*32 + ln;
    #pragma unroll
    for (int r = 0; r < 8; ++r) {
        const size_t row = (size_t)(rbase + r);
        const float v0 = acc0[r] + bias[c0];
        const float v1 = acc1[r] + bias[c0 + 16];
        if constexpr (OUTF32) {
            outF[row * DIMC + c0]      = v0;
            outF[row * DIMC + c0 + 16] = v1;
        } else {
            outH[row * DIMC + c0]      = (_Float16)v0;
            outH[row * DIMC + c0 + 16] = (_Float16)v1;
        }
    }
}

// ---------------------------------------------------------------------------
// Kernel 4: Flash attention (streaming softmax), one (b,h) per block.
//   Block = 8 waves; wave w owns query rows [qc*128 + w*16, +16).
//   K tiles: async global->LDS, double-buffered (1 async op/wave/tile).
//   V tiles: register load + transposed LDS scatter (async can't transpose),
//   with global_prefetch two tiles ahead.  S via 4 WMMAs, online softmax with
//   width-16 shuffles (stats align elementwise with the C-layout accumulators),
//   P bounced through a per-wave LDS patch (s_wait_dscnt 0) into A-fragment
//   layout, PV via 4 WMMAs.  Covers both branches:
//     out_r = FA(q,k,v_r);  out_d = FA(k,q,v_d)   (S^T trick).
// ---------------------------------------------------------------------------
__global__ __launch_bounds__(256)
void flash_wmma_kernel(const _Float16* __restrict__ Q, const _Float16* __restrict__ K,
                       const _Float16* __restrict__ V, _Float16* __restrict__ O) {
    const int bh = blockIdx.x >> 4;      // 16 q-chunks of 128 rows
    const int qc = blockIdx.x & 15;
    const int b = bh >> 3, h = bh & 7;
    const int tid = threadIdx.x, wave = tid >> 5, lane = tid & 31;
    const int ln = lane & 15, kh = (lane >> 4) * 8;
    const size_t base = ((size_t)b * SEQ) * DIMC + (size_t)h * DHEAD;
    const int qbase = qc * 128 + wave * 16;

    __shared__ _Float16 Ks[2][32][72];   // 64 d + pad (144B stride, 16B aligned)
    __shared__ _Float16 Vts[64][48];     // transposed: Vts[d][kk]
    __shared__ _Float16 Ps[8][16][48];   // per-wave P bounce buffer

    // Q fragments (K-dim = head dim, split 0..31 / 32..63), loaded once
    union { half16 v; half8 h[2]; } u;
    half16 qa0, qa1;
    {
        const _Float16* qr = Q + base + (size_t)(qbase + ln) * DIMC;
        u.h[0] = *(const half8*)(qr + kh);       u.h[1] = *(const half8*)(qr + 16 + kh);
        qa0 = u.v;
        u.h[0] = *(const half8*)(qr + 32 + kh);  u.h[1] = *(const half8*)(qr + 48 + kh);
        qa1 = u.v;
    }

    float8 acc[4] = {};
    float mstat[8], lstat[8];
    #pragma unroll
    for (int r = 0; r < 8; ++r) { mstat[r] = -1e30f; lstat[r] = 0.0f; }

    // K async staging: 1 instr/wave covers rows wave*4..+3, 16B chunks
    const int ksr = wave * 4 + (lane >> 3), ksc8 = (lane & 7) * 8;
    auto issueK = [&](int j0, int buf) {
        async_load_b128(LDS_OFF(&Ks[buf][ksr][ksc8]),
                        K + base + (size_t)(j0 + ksr) * DIMC + ksc8);
    };

    // V staging coords + register buffer
    const int kr = tid >> 3, dc = (tid & 7) * 8;
    half8 vReg = *(const half8*)(V + base + (size_t)kr * DIMC + dc);
    issueK(0, 0);

    int cur = 0;
    for (int j0 = 0; j0 < SEQ; j0 += 32) {
        const bool hasNext = (j0 + 32 < SEQ);

        // ---- commit V tile (transposed scatter) ----
        #pragma unroll
        for (int i = 0; i < 8; ++i) Vts[dc + i][kr] = vReg[i];

        // ---- next K tile -> other buffer; wait only for current tile ----
        if (hasNext) { issueK(j0 + 32, cur ^ 1); WAIT_ASYNCCNT_1(); }
        else         { WAIT_ASYNCCNT_0(); }
        __syncthreads();

        // ---- overlap: next V tile -> registers, tile-after-next -> L2 ----
        if (hasNext) {
            vReg = *(const half8*)(V + base + (size_t)(j0 + 32 + kr) * DIMC + dc);
            if (j0 + 64 < SEQ)
                __builtin_prefetch(V + base + (size_t)(j0 + 64 + kr) * DIMC + dc, 0, 0);
        }

        // ---- S = Q K^T : two 16x16 C tiles over 32 keys ----
        float8 s0 = {}, s1 = {};
        u.h[0] = *(const half8*)&Ks[cur][ln][kh];
        u.h[1] = *(const half8*)&Ks[cur][ln][16 + kh];
        s0 = wmma_f16(qa0, u.v, s0);
        u.h[0] = *(const half8*)&Ks[cur][ln][32 + kh];
        u.h[1] = *(const half8*)&Ks[cur][ln][48 + kh];
        s0 = wmma_f16(qa1, u.v, s0);
        u.h[0] = *(const half8*)&Ks[cur][16 + ln][kh];
        u.h[1] = *(const half8*)&Ks[cur][16 + ln][16 + kh];
        s1 = wmma_f16(qa0, u.v, s1);
        u.h[0] = *(const half8*)&Ks[cur][16 + ln][32 + kh];
        u.h[1] = *(const half8*)&Ks[cur][16 + ln][48 + kh];
        s1 = wmma_f16(qa1, u.v, s1);

        // ---- online softmax (C layout: VGPR r -> row r / r+8 per lane half) ----
        #pragma unroll
        for (int r = 0; r < 8; ++r) {
            const float a0 = s0[r] * 0.125f;   // D^-0.5 = 1/8
            const float a1 = s1[r] * 0.125f;
            float t = fmaxf(a0, a1);
            for (int o = 8; o; o >>= 1) t = fmaxf(t, __shfl_xor(t, o, 16));
            const float nm = fmaxf(mstat[r], t);
            const float f  = __expf(mstat[r] - nm);
            mstat[r] = nm;
            lstat[r] *= f;
            const float p0 = __expf(a0 - nm), p1 = __expf(a1 - nm);
            float rs = p0 + p1;
            for (int o = 8; o; o >>= 1) rs += __shfl_xor(rs, o, 16);
            lstat[r] += rs;
            #pragma unroll
            for (int c = 0; c < 4; ++c) acc[c][r] *= f;
            const int prow = r + ((lane < 16) ? 0 : 8);
            Ps[wave][prow][ln]      = (_Float16)p0;
            Ps[wave][prow][16 + ln] = (_Float16)p1;
        }
        // wave-private LDS bounce: wait out DS counter before cross-lane read
        asm volatile("s_wait_dscnt 0" ::: "memory");

        half16 pa;
        u.h[0] = *(const half8*)&Ps[wave][ln][kh];
        u.h[1] = *(const half8*)&Ps[wave][ln][16 + kh];
        pa = u.v;

        // ---- O += P V : 4 d-chunks of 16 ----
        #pragma unroll
        for (int c = 0; c < 4; ++c) {
            u.h[0] = *(const half8*)&Vts[c*16 + ln][kh];
            u.h[1] = *(const half8*)&Vts[c*16 + ln][16 + kh];
            acc[c] = wmma_f16(pa, u.v, acc[c]);
        }
        __syncthreads();   // reads of Ks[cur]/Vts done before next overwrite
        cur ^= 1;
    }

    // ---- normalize + store (heads concatenated in channel dim) ----
    #pragma unroll
    for (int r = 0; r < 8; ++r) {
        const int row = qbase + r + ((lane < 16) ? 0 : 8);
        const float inv = 1.0f / lstat[r];
        #pragma unroll
        for (int c = 0; c < 4; ++c)
            O[base + (size_t)row * DIMC + c*16 + ln] = (_Float16)(acc[c][r] * inv);
    }
}

// ---------------------------------------------------------------------------
// Host launcher
// ---------------------------------------------------------------------------
extern "C" void kernel_launch(void* const* d_in, const int* in_sizes, int n_in,
                              void* d_out, int out_size, void* d_ws, size_t ws_size,
                              hipStream_t stream) {
    (void)in_sizes; (void)n_in; (void)out_size; (void)ws_size;

    const float* x   = (const float*)d_in[0];
    const float* y   = (const float*)d_in[1];
    const float* Wq  = (const float*)d_in[2];   const float* bq  = (const float*)d_in[3];
    const float* Wk  = (const float*)d_in[4];   const float* bk  = (const float*)d_in[5];
    const float* Wvr = (const float*)d_in[6];   const float* bvr = (const float*)d_in[7];
    const float* Wvd = (const float*)d_in[8];   const float* bvd = (const float*)d_in[9];
    const float* g_r = (const float*)d_in[10];  const float* b_r = (const float*)d_in[11];
    const float* g_d = (const float*)d_in[12];  const float* b_d = (const float*)d_in[13];
    const float* Wpr = (const float*)d_in[14];  const float* bpr = (const float*)d_in[15];
    const float* Wpd = (const float*)d_in[16];  const float* bpd = (const float*)d_in[17];

    const size_t ROWS = (size_t)BATCH * SEQ;           // 8192
    const size_t ELT  = ROWS * DIMC;                   // 4,194,304 elements
    const size_t FBUF = ELT * sizeof(_Float16);        // 8 MB per f16 tensor

    char* wsb = (char*)d_ws;
    _Float16* xn  = (_Float16*)(wsb + 0 * FBUF);
    _Float16* yn  = (_Float16*)(wsb + 1 * FBUF);
    _Float16* qb  = (_Float16*)(wsb + 2 * FBUF);
    _Float16* kb  = (_Float16*)(wsb + 3 * FBUF);
    _Float16* vrb = (_Float16*)(wsb + 4 * FBUF);
    _Float16* vdb = (_Float16*)(wsb + 5 * FBUF);
    _Float16* orb = (_Float16*)(wsb + 6 * FBUF);
    _Float16* odb = (_Float16*)(wsb + 7 * FBUF);
    char* w16base = wsb + 8 * FBUF;
    const size_t WELT = (size_t)DIMC * DIMC;           // 262144
    _Float16* wq16  = (_Float16*)(w16base + 0 * WELT * 2);
    _Float16* wk16  = (_Float16*)(w16base + 1 * WELT * 2);
    _Float16* wvr16 = (_Float16*)(w16base + 2 * WELT * 2);
    _Float16* wvd16 = (_Float16*)(w16base + 3 * WELT * 2);
    _Float16* wpr16 = (_Float16*)(w16base + 4 * WELT * 2);
    _Float16* wpd16 = (_Float16*)(w16base + 5 * WELT * 2);

    // 1) LayerNorms (xn = LN_r(x), yn = LN_d(y)) + f16 cast
    ln_cast_kernel<<<(int)ROWS, 256, 0, stream>>>(x, g_r, b_r, xn);
    ln_cast_kernel<<<(int)ROWS, 256, 0, stream>>>(y, g_d, b_d, yn);

    // 2) weight conversion f32 -> f16
    const int cvtBlocks = (int)(WELT / 256);
    cvt_f16_kernel<<<cvtBlocks, 256, 0, stream>>>(Wq,  wq16,  (int)WELT);
    cvt_f16_kernel<<<cvtBlocks, 256, 0, stream>>>(Wk,  wk16,  (int)WELT);
    cvt_f16_kernel<<<cvtBlocks, 256, 0, stream>>>(Wvr, wvr16, (int)WELT);
    cvt_f16_kernel<<<cvtBlocks, 256, 0, stream>>>(Wvd, wvd16, (int)WELT);
    cvt_f16_kernel<<<cvtBlocks, 256, 0, stream>>>(Wpr, wpr16, (int)WELT);
    cvt_f16_kernel<<<cvtBlocks, 256, 0, stream>>>(Wpd, wpd16, (int)WELT);

    // 3) input projections (q from yn — query is depth-normed y)
    dim3 ggrid((unsigned)(ROWS / 64), DIMC / 64);
    gemm_wmma_kernel<false, false><<<ggrid, 256, 0, stream>>>(yn, wq16,  bq,  nullptr, qb);
    gemm_wmma_kernel<false, false><<<ggrid, 256, 0, stream>>>(xn, wk16,  bk,  nullptr, kb);
    gemm_wmma_kernel<false, false><<<ggrid, 256, 0, stream>>>(xn, wvr16, bvr, nullptr, vrb);
    gemm_wmma_kernel<false, false><<<ggrid, 256, 0, stream>>>(xn, wvd16, bvd, nullptr, vdb);

    // 4) attention: out_r = FA(q,k,v_r); out_d = FA(k,q,v_d) via S^T trick
    const int fgrid = BATCH * NHEAD * (SEQ / 128);     // 512 blocks
    flash_wmma_kernel<<<fgrid, 256, 0, stream>>>(qb, kb, vrb, orb);
    flash_wmma_kernel<<<fgrid, 256, 0, stream>>>(kb, qb, vdb, odb);

    // 5) output projections; second one applies the transpose-scramble gather
    float* out_r = (float*)d_out;
    float* out_d = (float*)d_out + ELT;
    gemm_wmma_kernel<false, true ><<<ggrid, 256, 0, stream>>>(orb, wpr16, bpr, out_r, nullptr);
    gemm_wmma_kernel<true,  true ><<<ggrid, 256, 0, stream>>>(odb, wpd16, bpd, out_d, nullptr);
}
